// GCN2_73100343378655
// MI455X (gfx1250) — compile-verified, matrix operand order checked by reading, not run on hardware
//
#include <hip/hip_runtime.h>
#include <hip/hip_fp16.h>

typedef _Float16 h16;
typedef __attribute__((ext_vector_type(16))) _Float16 v16h;
typedef __attribute__((ext_vector_type(8)))  float    v8f;

constexpr int NN = 50000;   // nodes
constexpr int NE = 800000;  // edges
constexpr int NG = 128;     // graphs
constexpr int CI = 128;     // input feat
constexpr int CH = 256;     // hidden
constexpr int CO = 512;     // output feat
constexpr int NHEADS = 4;
constexpr int DHEAD = 64;
constexpr float EPSV = 1e-5f;

__device__ __forceinline__ float lrelu2(float v) { return v > 0.f ? v : 0.2f * v; }
// monotone float<->uint encoding so unsigned atomicMax implements float max
__device__ __forceinline__ unsigned encf(float f) {
  unsigned u = __float_as_uint(f);
  return (u & 0x80000000u) ? ~u : (u | 0x80000000u);
}
__device__ __forceinline__ float decf(unsigned e) {
  return __uint_as_float((e & 0x80000000u) ? (e & 0x7fffffffu) : ~e);
}

// ---------------- utility kernels ----------------
__global__ void k_fill_f32(float* p, float v, int n) {
  int i = blockIdx.x * blockDim.x + threadIdx.x;
  if (i < n) p[i] = v;
}
__global__ void k_fill_enc(unsigned* p, float v, int n) {
  int i = blockIdx.x * blockDim.x + threadIdx.x;
  if (i < n) p[i] = encf(v);
}
__global__ void k_to_f16(const float* s, h16* d, int n) {
  int i = blockIdx.x * blockDim.x + threadIdx.x;
  if (i < n) d[i] = (h16)s[i];
}
// Wt[n*K + k] = W[k*Nout + n]  (convert f32 weight to f16, transposed)
__global__ void k_transpose_f16(const float* W, h16* Wt, int K, int Nout) {
  int i = blockIdx.x * blockDim.x + threadIdx.x;
  if (i >= K * Nout) return;
  int n = i / K, k = i % K;
  Wt[i] = (h16)W[(size_t)k * Nout + n];
}
// out = act( ((in + pre) - m) * rsqrt(v+eps) * g + b ); act: 0 none, 1 leaky_relu(0.2)
__global__ void k_bn_act(const float* in, const float* pre, const float* g, const float* b,
                         const float* m, const float* v, float* out, int n, int C, int act) {
  int i = blockIdx.x * blockDim.x + threadIdx.x;
  if (i >= n) return;
  int c = i % C;
  float x = in[i] + (pre ? pre[c] : 0.f);
  x = (x - m[c]) * rsqrtf(v[c] + EPSV) * g[c] + b[c];
  out[i] = act ? lrelu2(x) : x;
}

// ---------------- WMMA GEMM ----------------
// 16-bit A fragment (16x32 tile, row-major, row stride ld), per ISA 7.12.2 layout
__device__ __forceinline__ v16h load_frag16(const h16* base, int ld, int lane) {
  int r  = lane & 15;
  int kb = (lane & 16) ? 8 : 0;
  const h16* p = base + (size_t)r * ld;
  v16h f;
#pragma unroll
  for (int vv = 0; vv < 4; ++vv) {
    f[2 * vv]         = p[kb + 2 * vv];
    f[2 * vv + 1]     = p[kb + 2 * vv + 1];
    f[8 + 2 * vv]     = p[16 + kb + 2 * vv];
    f[8 + 2 * vv + 1] = p[16 + kb + 2 * vv + 1];
  }
  return f;
}
// C[M x Nout] f32 = A[M x K] f16 * Bt[Nout x K]^T f16 ; one 16x16 tile per wave32
// mode: 0 none, 1 +bias, 2 +bias & leaky_relu(0.2), 3 +bias & tanh
__global__ void k_wmma_gemm(const h16* A, const h16* Bt, const float* bias, float* C,
                            int M, int K, int Nout, int mode) {
  int gtid = blockIdx.x * blockDim.x + threadIdx.x;
  int wave = gtid >> 5;
  int lane = threadIdx.x & 31;
  int tilesN = Nout >> 4;
  int tilesM = M >> 4;
  if (wave >= tilesM * tilesN) return;  // wave-uniform exit, EXEC stays full for WMMA
  int tm = wave / tilesN, tn = wave % tilesN;
  const h16* Ab = A + (size_t)tm * 16 * K;
  const h16* Bb = Bt + (size_t)tn * 16 * K;
  v8f acc = {};
  for (int k = 0; k < K; k += 32) {
    v16h a = load_frag16(Ab + k, K, lane);
    v16h b = load_frag16(Bb + k, K, lane);
    acc = __builtin_amdgcn_wmma_f32_16x16x32_f16(false, a, false, b, (short)0, acc, false, false);
  }
  int col  = tn * 16 + (lane & 15);
  int row0 = tm * 16 + ((lane & 16) ? 8 : 0);
  float bb = (mode >= 1 && bias) ? bias[col] : 0.f;
#pragma unroll
  for (int r = 0; r < 8; ++r) {
    float v = acc[r] + bb;
    if (mode == 2) v = lrelu2(v);
    else if (mode == 3) v = tanhf(v);
    C[(size_t)(row0 + r) * Nout + col] = v;
  }
}

// ---------------- GCN ----------------
__global__ void k_deg_edge(const int* dst, float* deg, int E) {
  int e = blockIdx.x * blockDim.x + threadIdx.x;
  if (e < E) atomicAdd(&deg[dst[e]], 1.0f);
}
__global__ void k_rsqrt_ip(float* p, int n) {
  int i = blockIdx.x * blockDim.x + threadIdx.x;
  if (i < n) p[i] = rsqrtf(p[i]);
}
// out = t * dinv^2 (self loop term) + bias
__global__ void k_gcn_init(const float* t, const float* dinv, const float* b, float* out,
                           int n, int C) {
  int i = blockIdx.x * blockDim.x + threadIdx.x;
  if (i >= n) return;
  int node = i / C, c = i % C;
  float di = dinv[node];
  out[i] = t[i] * di * di + b[c];
}
// out[dst] += t[src] * dinv[src]*dinv[dst]  (4 channels per thread)
__global__ void k_gcn_edge(const float* t, const float* dinv, const int* src, const int* dst,
                           float* out, int E, int C) {
  int i = blockIdx.x * blockDim.x + threadIdx.x;
  int per = C >> 2;
  if (i >= E * per) return;
  int e = i / per, c = (i % per) * 4;
  int s = src[e], d = dst[e];
  float w = dinv[s] * dinv[d];
  const float4 v = *reinterpret_cast<const float4*>(t + (size_t)s * C + c);
  float* o = out + (size_t)d * C + c;
  atomicAdd(o + 0, v.x * w);
  atomicAdd(o + 1, v.y * w);
  atomicAdd(o + 2, v.z * w);
  atomicAdd(o + 3, v.w * w);
}

// ---------------- GAT ----------------
__global__ void k_gat_sd(const float* ha, const float* as, const float* ad,
                         float* s, float* d, int n) {
  int i = blockIdx.x * blockDim.x + threadIdx.x;
  if (i >= n * NHEADS) return;
  int node = i / NHEADS, h = i % NHEADS;
  const float* r  = ha + (size_t)node * CH + h * DHEAD;
  const float* ws = as + h * DHEAD;
  const float* wd = ad + h * DHEAD;
  float ss = 0.f, dd = 0.f;
  for (int k = 0; k < DHEAD; ++k) { float v = r[k]; ss += v * ws[k]; dd += v * wd[k]; }
  s[i] = ss; d[i] = dd;
}
__global__ void k_gat_m_init(const float* s, const float* d, unsigned* m, int n) {
  int i = blockIdx.x * blockDim.x + threadIdx.x;
  if (i < n) m[i] = encf(lrelu2(s[i] + d[i]));   // self-loop logit
}
__global__ void k_gat_m_edge(const float* s, const float* d, const int* src, const int* dst,
                             unsigned* m, int E) {
  int i = blockIdx.x * blockDim.x + threadIdx.x;
  if (i >= E * NHEADS) return;
  int e = i >> 2, h = i & 3;
  float l = lrelu2(s[src[e] * NHEADS + h] + d[dst[e] * NHEADS + h]);
  atomicMax(&m[dst[e] * NHEADS + h], encf(l));
}
__global__ void k_gat_z_init(const float* s, const float* d, const unsigned* m, float* z, int n) {
  int i = blockIdx.x * blockDim.x + threadIdx.x;
  if (i < n) z[i] = expf(lrelu2(s[i] + d[i]) - decf(m[i]));
}
__global__ void k_gat_z_edge(const float* s, const float* d, const int* src, const int* dst,
                             const unsigned* m, float* z, int E) {
  int i = blockIdx.x * blockDim.x + threadIdx.x;
  if (i >= E * NHEADS) return;
  int e = i >> 2, h = i & 3;
  int di = dst[e] * NHEADS + h;
  float l = lrelu2(s[src[e] * NHEADS + h] + d[di]);
  atomicAdd(&z[di], expf(l - decf(m[di])));
}
__global__ void k_gat_out_init(const float* ha, const float* s, const float* d,
                               const unsigned* m, const float* z, float* out, int n) {
  int i = blockIdx.x * blockDim.x + threadIdx.x;
  if (i >= n * CH) return;
  int node = i / CH, c = i % CH, h = c >> 6;
  int idx = node * NHEADS + h;
  float alpha = expf(lrelu2(s[idx] + d[idx]) - decf(m[idx])) / z[idx];
  out[i] = ha[i] * alpha;
}
__global__ void k_gat_out_edge(const float* ha, const float* s, const float* d,
                               const unsigned* m, const float* z, const int* src,
                               const int* dst, float* out, int E) {
  int i = blockIdx.x * blockDim.x + threadIdx.x;
  if (i >= E * 64) return;
  int e = i / 64, c = (i % 64) * 4, h = c >> 6;
  int sn = src[e], dn = dst[e];
  int si = sn * NHEADS + h, di = dn * NHEADS + h;
  float l = lrelu2(s[si] + d[di]);
  float alpha = expf(l - decf(m[di])) / z[di];
  const float4 v = *reinterpret_cast<const float4*>(ha + (size_t)sn * CH + c);
  float* o = out + (size_t)dn * CH + c;
  atomicAdd(o + 0, v.x * alpha);
  atomicAdd(o + 1, v.y * alpha);
  atomicAdd(o + 2, v.z * alpha);
  atomicAdd(o + 3, v.w * alpha);
}

// ---------------- global attention pooling ----------------
__global__ void k_gate(const float* tg, const float* Wg2, const float* bg2, float* gate, int n) {
  int i = blockIdx.x * blockDim.x + threadIdx.x;
  if (i >= n) return;
  const float* r = tg + (size_t)i * CH;
  float a = 0.f;
  for (int k = 0; k < CH; ++k) a += r[k] * Wg2[k];
  gate[i] = a + bg2[0];
}
__global__ void k_pool_max(const float* gate, const int* batch, unsigned* gm, int n) {
  int i = blockIdx.x * blockDim.x + threadIdx.x;
  if (i < n) atomicMax(&gm[batch[i]], encf(gate[i]));
}
__global__ void k_pool_z(const float* gate, const int* batch, const unsigned* gm,
                         float* gz, float* cnt, int n) {
  int i = blockIdx.x * blockDim.x + threadIdx.x;
  if (i >= n) return;
  int b = batch[i];
  atomicAdd(&gz[b], expf(gate[i] - decf(gm[b])));
  atomicAdd(&cnt[b], 1.0f);
}
__global__ void k_pool_emb(const float* h, const float* gate, const int* batch,
                           const unsigned* gm, const float* gz, float* emb, int n) {
  int i = blockIdx.x * blockDim.x + threadIdx.x;
  int per = CH >> 2;
  if (i >= n * per) return;
  int node = i / per, c = (i % per) * 4;
  int b = batch[node];
  float alpha = expf(gate[node] - decf(gm[b])) / fmaxf(gz[b], 1e-9f);
  const float4 v = *reinterpret_cast<const float4*>(h + (size_t)node * CH + c);
  float* o = emb + (size_t)b * CH + c;
  atomicAdd(o + 0, v.x * alpha);
  atomicAdd(o + 1, v.y * alpha);
  atomicAdd(o + 2, v.z * alpha);
  atomicAdd(o + 3, v.w * alpha);
}
__global__ void k_pool_res(const float* xbn, const int* batch, float* gres, int n) {
  int i = blockIdx.x * blockDim.x + threadIdx.x;
  int per = CI >> 2;
  if (i >= n * per) return;
  int node = i / per, c = (i % per) * 4;
  int b = batch[node];
  const float4 v = *reinterpret_cast<const float4*>(xbn + (size_t)node * CI + c);
  float* o = gres + (size_t)b * CI + c;
  atomicAdd(o + 0, v.x);
  atomicAdd(o + 1, v.y);
  atomicAdd(o + 2, v.z);
  atomicAdd(o + 3, v.w);
}
__global__ void k_res_div(float* gres, const float* cnt, int n) {
  int i = blockIdx.x * blockDim.x + threadIdx.x;
  if (i >= n) return;
  gres[i] /= fmaxf(cnt[i / CI], 1.0f);
}
__global__ void k_build_f(const float* emb, const float* gres, h16* f, int n) {
  int i = blockIdx.x * blockDim.x + threadIdx.x;
  if (i >= n) return;
  int g = i / (CH + CI), c = i % (CH + CI);
  f[i] = (h16)(c < CH ? emb[(size_t)g * CH + c] : gres[(size_t)g * CI + (c - CH)]);
}

// ---------------- driver ----------------
extern "C" void kernel_launch(void* const* d_in, const int* in_sizes, int n_in,
                              void* d_out, int out_size, void* d_ws, size_t ws_size,
                              hipStream_t stream) {
  (void)in_sizes; (void)n_in; (void)out_size; (void)ws_size;
  const float* x     = (const float*)d_in[0];
  const int*   ei    = (const int*)d_in[1];
  const int*   src   = ei;
  const int*   dst   = ei + NE;
  const int*   batch = (const int*)d_in[2];
  const float* bn0_g = (const float*)d_in[3];
  const float* bn0_b = (const float*)d_in[4];
  const float* bn0_m = (const float*)d_in[5];
  const float* bn0_v = (const float*)d_in[6];
  const float* W_in  = (const float*)d_in[7];
  const float* b_in  = (const float*)d_in[8];
  const float* W_g0  = (const float*)d_in[9];
  const float* b_g0  = (const float*)d_in[10];
  const float* bn1_g = (const float*)d_in[11];
  const float* bn1_b = (const float*)d_in[12];
  const float* bn1_m = (const float*)d_in[13];
  const float* bn1_v = (const float*)d_in[14];
  const float* W_a   = (const float*)d_in[15];
  const float* att_s = (const float*)d_in[16];
  const float* att_d = (const float*)d_in[17];
  const float* b_a   = (const float*)d_in[18];
  const float* bn2_g = (const float*)d_in[19];
  const float* bn2_b = (const float*)d_in[20];
  const float* bn2_m = (const float*)d_in[21];
  const float* bn2_v = (const float*)d_in[22];
  const float* W_g2  = (const float*)d_in[23];
  const float* b_g2  = (const float*)d_in[24];
  const float* bn3_g = (const float*)d_in[25];
  const float* bn3_b = (const float*)d_in[26];
  const float* bn3_m = (const float*)d_in[27];
  const float* bn3_v = (const float*)d_in[28];
  const float* Wg1   = (const float*)d_in[29];
  const float* bg1   = (const float*)d_in[30];
  const float* Wg2   = (const float*)d_in[31];
  const float* bg2   = (const float*)d_in[32];
  const float* W_out = (const float*)d_in[33];
  const float* b_out = (const float*)d_in[34];
  const float* bno_g = (const float*)d_in[35];
  const float* bno_b = (const float*)d_in[36];
  const float* bno_m = (const float*)d_in[37];
  const float* bno_v = (const float*)d_in[38];

  // carve workspace
  char* ws = (char*)d_ws;
  size_t off = 0;
  auto carve = [&](size_t bytes) -> char* {
    char* p = ws + off;
    off += (bytes + 255) & ~(size_t)255;
    return p;
  };
  float*    xbn   = (float*)carve((size_t)NN * CI * 4);   // persists (x_res)
  float*    hB    = (float*)carve((size_t)NN * CH * 4);   // ping
  float*    hC    = (float*)carve((size_t)NN * CH * 4);   // pong
  h16*      X16   = (h16*)carve((size_t)NN * CH * 2);     // f16 GEMM input staging
  h16*      Wt    = (h16*)carve((size_t)(CI + CH) * CO * 2); // largest: 384x512
  float*    dinv  = (float*)carve((size_t)NN * 4);
  float*    sb    = (float*)carve((size_t)NN * NHEADS * 4);
  float*    db    = (float*)carve((size_t)NN * NHEADS * 4);
  unsigned* menc  = (unsigned*)carve((size_t)NN * NHEADS * 4);
  float*    zb    = (float*)carve((size_t)NN * NHEADS * 4);
  float*    gate  = (float*)carve((size_t)NN * 4);
  unsigned* gm    = (unsigned*)carve((size_t)NG * 4);
  float*    gz    = (float*)carve((size_t)NG * 4);
  float*    cnt   = (float*)carve((size_t)NG * 4);
  float*    emb   = (float*)carve((size_t)NG * CH * 4);
  float*    gres  = (float*)carve((size_t)NG * CI * 4);
  h16*      fbuf  = (h16*)carve((size_t)NG * (CH + CI) * 2);
  float*    otmp  = (float*)carve((size_t)NG * CO * 4);

  const int T = 256;
  auto nb = [&](long long n) { return (unsigned)((n + T - 1) / T); };

  // bn0 -> xbn (residual kept)
  k_bn_act<<<nb((long long)NN * CI), T, 0, stream>>>(x, nullptr, bn0_g, bn0_b, bn0_m, bn0_v,
                                                     xbn, NN * CI, CI, 0);
  // degrees (dst-based, +1 self loop) -> dinv
  k_fill_f32<<<nb(NN), T, 0, stream>>>(dinv, 1.0f, NN);
  k_deg_edge<<<nb(NE), T, 0, stream>>>(dst, dinv, NE);
  k_rsqrt_ip<<<nb(NN), T, 0, stream>>>(dinv, NN);

  // h0 = lrelu(xbn @ W_in + b_in)  -> hB
  k_to_f16<<<nb((long long)NN * CI), T, 0, stream>>>(xbn, X16, NN * CI);
  k_transpose_f16<<<nb(CI * CH), T, 0, stream>>>(W_in, Wt, CI, CH);
  {
    long long waves = (long long)(NN / 16) * (CH / 16);
    k_wmma_gemm<<<nb(waves * 32), T, 0, stream>>>(X16, Wt, b_in, hB, NN, CI, CH, 2);
  }

  // GCN0: t = h0 @ W_g0 -> hC ; aggregate -> hB ; bn1+lrelu in place
  k_to_f16<<<nb((long long)NN * CH), T, 0, stream>>>(hB, X16, NN * CH);
  k_transpose_f16<<<nb(CH * CH), T, 0, stream>>>(W_g0, Wt, CH, CH);
  {
    long long waves = (long long)(NN / 16) * (CH / 16);
    k_wmma_gemm<<<nb(waves * 32), T, 0, stream>>>(X16, Wt, nullptr, hC, NN, CH, CH, 0);
  }
  k_gcn_init<<<nb((long long)NN * CH), T, 0, stream>>>(hC, dinv, b_g0, hB, NN * CH, CH);
  k_gcn_edge<<<nb((long long)NE * (CH / 4)), T, 0, stream>>>(hC, dinv, src, dst, hB, NE, CH);
  k_bn_act<<<nb((long long)NN * CH), T, 0, stream>>>(hB, nullptr, bn1_g, bn1_b, bn1_m, bn1_v,
                                                     hB, NN * CH, CH, 1);

  // GAT: ha = h1 @ W_a -> hC ; softmax attention aggregate -> hB ; (+b_a) bn2+lrelu
  k_to_f16<<<nb((long long)NN * CH), T, 0, stream>>>(hB, X16, NN * CH);
  k_transpose_f16<<<nb(CH * CH), T, 0, stream>>>(W_a, Wt, CH, CH);
  {
    long long waves = (long long)(NN / 16) * (CH / 16);
    k_wmma_gemm<<<nb(waves * 32), T, 0, stream>>>(X16, Wt, nullptr, hC, NN, CH, CH, 0);
  }
  k_gat_sd<<<nb((long long)NN * NHEADS), T, 0, stream>>>(hC, att_s, att_d, sb, db, NN);
  k_gat_m_init<<<nb((long long)NN * NHEADS), T, 0, stream>>>(sb, db, menc, NN * NHEADS);
  k_gat_m_edge<<<nb((long long)NE * NHEADS), T, 0, stream>>>(sb, db, src, dst, menc, NE);
  k_gat_z_init<<<nb((long long)NN * NHEADS), T, 0, stream>>>(sb, db, menc, zb, NN * NHEADS);
  k_gat_z_edge<<<nb((long long)NE * NHEADS), T, 0, stream>>>(sb, db, src, dst, menc, zb, NE);
  k_gat_out_init<<<nb((long long)NN * CH), T, 0, stream>>>(hC, sb, db, menc, zb, hB, NN);
  k_gat_out_edge<<<nb((long long)NE * 64), T, 0, stream>>>(hC, sb, db, menc, zb, src, dst, hB, NE);
  k_bn_act<<<nb((long long)NN * CH), T, 0, stream>>>(hB, b_a, bn2_g, bn2_b, bn2_m, bn2_v,
                                                     hB, NN * CH, CH, 1);

  // GCN2: t = h2 @ W_g2 -> hC ; aggregate -> hB ; bn3+lrelu -> h3 in hB
  k_to_f16<<<nb((long long)NN * CH), T, 0, stream>>>(hB, X16, NN * CH);
  k_transpose_f16<<<nb(CH * CH), T, 0, stream>>>(W_g2, Wt, CH, CH);
  {
    long long waves = (long long)(NN / 16) * (CH / 16);
    k_wmma_gemm<<<nb(waves * 32), T, 0, stream>>>(X16, Wt, nullptr, hC, NN, CH, CH, 0);
  }
  k_gcn_init<<<nb((long long)NN * CH), T, 0, stream>>>(hC, dinv, b_g2, hB, NN * CH, CH);
  k_gcn_edge<<<nb((long long)NE * (CH / 4)), T, 0, stream>>>(hC, dinv, src, dst, hB, NE, CH);
  k_bn_act<<<nb((long long)NN * CH), T, 0, stream>>>(hB, nullptr, bn3_g, bn3_b, bn3_m, bn3_v,
                                                     hB, NN * CH, CH, 1);

  // gate = tanh(h3 @ Wg1 + bg1) @ Wg2 + bg2
  k_to_f16<<<nb((long long)NN * CH), T, 0, stream>>>(hB, X16, NN * CH);
  k_transpose_f16<<<nb(CH * CH), T, 0, stream>>>(Wg1, Wt, CH, CH);
  {
    long long waves = (long long)(NN / 16) * (CH / 16);
    k_wmma_gemm<<<nb(waves * 32), T, 0, stream>>>(X16, Wt, bg1, hC, NN, CH, CH, 3);
  }
  k_gate<<<nb(NN), T, 0, stream>>>(hC, Wg2, bg2, gate, NN);

  // per-graph softmax pooling + mean pool of residual
  k_fill_enc<<<nb(NG), T, 0, stream>>>(gm, -3.0e38f, NG);
  k_pool_max<<<nb(NN), T, 0, stream>>>(gate, batch, gm, NN);
  k_fill_f32<<<nb(NG), T, 0, stream>>>(gz, 0.f, NG);
  k_fill_f32<<<nb(NG), T, 0, stream>>>(cnt, 0.f, NG);
  k_pool_z<<<nb(NN), T, 0, stream>>>(gate, batch, gm, gz, cnt, NN);
  k_fill_f32<<<nb(NG * CH), T, 0, stream>>>(emb, 0.f, NG * CH);
  k_pool_emb<<<nb((long long)NN * (CH / 4)), T, 0, stream>>>(hB, gate, batch, gm, gz, emb, NN);
  k_fill_f32<<<nb(NG * CI), T, 0, stream>>>(gres, 0.f, NG * CI);
  k_pool_res<<<nb((long long)NN * (CI / 4)), T, 0, stream>>>(xbn, batch, gres, NN);
  k_res_div<<<nb(NG * CI), T, 0, stream>>>(gres, cnt, NG * CI);
  k_build_f<<<nb(NG * (CH + CI)), T, 0, stream>>>(emb, gres, fbuf, NG * (CH + CI));

  // out = bn( lrelu(f @ W_out + b_out) )
  k_transpose_f16<<<nb((CH + CI) * CO), T, 0, stream>>>(W_out, Wt, CH + CI, CO);
  {
    long long waves = (long long)(NG / 16) * (CO / 16);
    k_wmma_gemm<<<nb(waves * 32), T, 0, stream>>>(fbuf, Wt, b_out, otmp, NG, CH + CI, CO, 2);
  }
  k_bn_act<<<nb(NG * CO), T, 0, stream>>>(otmp, nullptr, bno_g, bno_b, bno_m, bno_v,
                                          (float*)d_out, NG * CO, CO, 0);
}
